// DynamicEdgeConv_v1_69655779606947
// MI455X (gfx1250) — matrix operand order repaired and checked by reading, not compiled, not run on hardware
//
#include <hip/hip_runtime.h>
#include <hip/hip_bf16.h>

#define K_NBR 8
#define PB    1024
#define BG    16
#define NPTS  (BG * PB)        // 16384 points
#define MEDGE (NPTS * K_NBR)   // 131072 edge rows
#define BN_EPS 1e-5f

typedef __attribute__((ext_vector_type(16))) _Float16 v16h;
typedef __attribute__((ext_vector_type(8)))  float    v8f;
typedef __attribute__((ext_vector_type(8)))  _Float16 h8;

static __device__ __forceinline__ h8 ld8(const _Float16* p) {
    return *(const h8*)p;
}

// ---------------------------------------------------------------------------
// small utility kernels
// ---------------------------------------------------------------------------
__global__ void f32_to_f16_kernel(const float* __restrict__ in,
                                  _Float16* __restrict__ out, int n) {
    int i = blockIdx.x * blockDim.x + threadIdx.x;
    if (i < n) out[i] = (_Float16)in[i];
}

__global__ void copy_x_kernel(const float* __restrict__ x,
                              float* __restrict__ xc,
                              _Float16* __restrict__ xh, int n) {
    int i = blockIdx.x * blockDim.x + threadIdx.x;
    if (i < n) { float v = x[i]; xc[i] = v; xh[i] = (_Float16)v; }
}

// ---------------------------------------------------------------------------
// kNN: one wave (32 lanes) per point; per-lane top-8 insertion, LDS merge
// ---------------------------------------------------------------------------
__global__ void knn_kernel(const float* __restrict__ x,
                           int* __restrict__ knn, int cin) {
    __shared__ float xi[256];
    __shared__ float cd[32 * K_NBR];
    __shared__ int   cj[32 * K_NBR];
    const int pt   = blockIdx.x;        // 0..16383
    const int b    = pt >> 10;
    const int i    = pt & (PB - 1);
    const int lane = threadIdx.x;       // 0..31
    const float* xb = x + (size_t)b * PB * cin;

    for (int c = lane; c < cin; c += 32) xi[c] = xb[(size_t)i * cin + c];
    __syncthreads();

    float bd[K_NBR]; int bj[K_NBR];
#pragma unroll
    for (int s = 0; s < K_NBR; ++s) { bd[s] = 3.4e38f; bj[s] = 0; }

    const bool vec4 = ((cin & 3) == 0);
    for (int j = lane; j < PB; j += 32) {
        const float* xj = xb + (size_t)j * cin;
        __builtin_prefetch(xb + (size_t)(j + 32) * cin, 0, 0);
        float d = 0.f;
        if (vec4) {
            const float4* xj4 = (const float4*)xj;
            for (int c4 = 0; c4 < (cin >> 2); ++c4) {
                float4 v = xj4[c4];
                float t0 = xi[c4 * 4 + 0] - v.x;
                float t1 = xi[c4 * 4 + 1] - v.y;
                float t2 = xi[c4 * 4 + 2] - v.z;
                float t3 = xi[c4 * 4 + 3] - v.w;
                d += t0 * t0 + t1 * t1 + t2 * t2 + t3 * t3;
            }
        } else {
            for (int c = 0; c < cin; ++c) { float t = xi[c] - xj[c]; d += t * t; }
        }
        if (d < bd[K_NBR - 1]) {
            int s = K_NBR - 1;
            while (s > 0 && bd[s - 1] > d) { bd[s] = bd[s - 1]; bj[s] = bj[s - 1]; --s; }
            bd[s] = d; bj[s] = j;
        }
    }
#pragma unroll
    for (int s = 0; s < K_NBR; ++s) {
        cd[lane * K_NBR + s] = bd[s];
        cj[lane * K_NBR + s] = bj[s];
    }
    __syncthreads();
    if (lane == 0) {
        for (int s = 0; s < K_NBR; ++s) {
            float best = 3.5e38f; int bi = 0;
            for (int t = 0; t < 32 * K_NBR; ++t)
                if (cd[t] < best) { best = cd[t]; bi = t; }
            knn[(size_t)pt * K_NBR + s] = cj[bi];
            cd[bi] = 3.6e38f;
        }
    }
}

// ---------------------------------------------------------------------------
// GEMM1 (fused edge-feature gather): H1 = [xi, xn-xi] @ W1^T + b1  -> f16
// Templated so region tests are wave-uniform scalar branches (CIN%32==0) and
// all fragment loads are contiguous 16B h8 loads.
// Fragment maps (per ISA tables):
//  A: lane&15 = M, half t: K = kb + t + (t>=8?8:0) + (lane>=16?8:0)
//     -> two contiguous 8-half chunks at kb+ahi and kb+16+ahi
//  B: lane&15 = N, half t: K = kb + t + (t>=8?8:0) + (lane>=16?8:0)
//     -> two contiguous 8-half chunks at kb+ahi and kb+16+ahi
//  C/D: vgpr r -> M = r + (lane>=16?8:0), N = lane&15
// ---------------------------------------------------------------------------
template<int CIN, int COUT>
__global__ void gemm1_edge_kernel(const _Float16* __restrict__ xh,
                                  const int* __restrict__ knn,
                                  const _Float16* __restrict__ w1h,
                                  const float* __restrict__ b1f,
                                  _Float16* __restrict__ h1h) {
    constexpr int K2 = 2 * CIN;
    constexpr int Kp = (K2 + 31) & ~31;
    const int lane = threadIdx.x & 31;
    const int wv   = threadIdx.x >> 5;
    const int tm = blockIdx.x * 4 + wv;
    const int tn = blockIdx.y;
    const int l15 = lane & 15;
    const int ahi = (lane >= 16) ? 8 : 0;
    const int bhi = (lane >= 16) ? 16 : 0;

    const int mrow = tm * 16 + l15;
    const int kk = mrow & 7;
    const int pi = (mrow >> 3) & (PB - 1);
    const int b  = mrow >> 13;
    const int nbr = knn[(size_t)(b * PB + pi) * K_NBR + kk];
    const _Float16* xb = xh + (size_t)b * PB * CIN;
    const _Float16* xi = xb + (size_t)pi  * CIN;
    const _Float16* xn = xb + (size_t)nbr * CIN;

    const int ncol = tn * 16 + l15;
    const _Float16* wn = w1h + (size_t)ncol * K2;

    v8f acc = {};
    for (int kb = 0; kb < Kp; kb += 32) {
        v16h a, bb;
        if constexpr ((CIN % 32) == 0) {
            h8 alo, ahiv;
            if (kb < CIN) {               // wave-uniform: pure x_i region
                alo  = ld8(xi + kb + ahi);
                ahiv = ld8(xi + kb + 16 + ahi);
            } else {                      // wave-uniform: (x_n - x_i) region
                const int c0 = kb - CIN;
                alo  = ld8(xn + c0 + ahi)      - ld8(xi + c0 + ahi);
                ahiv = ld8(xn + c0 + 16 + ahi) - ld8(xi + c0 + 16 + ahi);
            }
            a = __builtin_shufflevector(alo, ahiv,
                0, 1, 2, 3, 4, 5, 6, 7, 8, 9, 10, 11, 12, 13, 14, 15);
            h8 blo = ld8(wn + kb + ahi);
            h8 bhh = ld8(wn + kb + 16 + ahi);
            bb = __builtin_shufflevector(blo, bhh,
                0, 1, 2, 3, 4, 5, 6, 7, 8, 9, 10, 11, 12, 13, 14, 15);
        } else {
            // generic path (layer 0 only: CIN=5, single k-step)
#pragma unroll
            for (int t = 0; t < 16; ++t) {
                int k = kb + t + ((t >= 8) ? 8 : 0) + ahi;
                float v = 0.f;
                if (k < CIN)      v = (float)xi[k];
                else if (k < K2)  { int c = k - CIN; v = (float)xn[c] - (float)xi[c]; }
                a[t] = (_Float16)v;
                int k2 = kb + t + ((t >= 8) ? 8 : 0) + ahi;
                bb[t] = (k2 < K2) ? wn[k2] : (_Float16)0.f;
            }
        }
        acc = __builtin_amdgcn_wmma_f32_16x16x32_f16(
            false, a, false, bb, (short)0, acc, false, false);
    }
    const float bias = b1f[ncol];
#pragma unroll
    for (int r = 0; r < 8; ++r) {
        int mg = tm * 16 + r + ahi;
        h1h[(size_t)mg * COUT + ncol] = (_Float16)(acc[r] + bias);
    }
}

// BN1 affine + ReLU applied in-place to h1 (f16) so GEMM2 does pure loads
__global__ void bn1_apply_kernel(_Float16* __restrict__ h,
                                 const float* __restrict__ sc,
                                 const float* __restrict__ sh,
                                 int cout, int n) {
    int idx = blockIdx.x * blockDim.x + threadIdx.x;
    if (idx >= n) return;
    int c = idx % cout;
    float v = sc[c] * (float)h[idx] + sh[c];
    h[idx] = (_Float16)(v > 0.f ? v : 0.f);
}

// ---------------------------------------------------------------------------
// GEMM2: D = relu_bn1(H1) @ W2^T; the 16 tile rows are exactly two K=8 edge
// groups -> deterministic in-wave sum over K, + 8*b2, store to s[B*P, cout].
// ---------------------------------------------------------------------------
template<int COUT>
__global__ void gemm2_fused_kernel(const _Float16* __restrict__ h1h,
                                   const _Float16* __restrict__ w2h,
                                   const float* __restrict__ b2f,
                                   float* __restrict__ sbuf) {
    const int lane = threadIdx.x & 31;
    const int wv   = threadIdx.x >> 5;
    const int tm = blockIdx.x * 4 + wv;
    const int tn = blockIdx.y;
    const int l15 = lane & 15;
    const int ahi = (lane >= 16) ? 8 : 0;
    const int bhi = (lane >= 16) ? 16 : 0;

    const int mrow = tm * 16 + l15;
    const _Float16* arow = h1h + (size_t)mrow * COUT;
    const int ncol = tn * 16 + l15;
    const _Float16* wn = w2h + (size_t)ncol * COUT;

    v8f acc = {};
#pragma unroll 2
    for (int kb = 0; kb < COUT; kb += 32) {
        h8 alo = ld8(arow + kb + ahi);
        h8 ahiv = ld8(arow + kb + 16 + ahi);
        v16h a = __builtin_shufflevector(alo, ahiv,
            0, 1, 2, 3, 4, 5, 6, 7, 8, 9, 10, 11, 12, 13, 14, 15);
        h8 blo = ld8(wn + kb + ahi);
        h8 bhh = ld8(wn + kb + 16 + ahi);
        v16h bb = __builtin_shufflevector(blo, bhh,
            0, 1, 2, 3, 4, 5, 6, 7, 8, 9, 10, 11, 12, 13, 14, 15);
        acc = __builtin_amdgcn_wmma_f32_16x16x32_f16(
            false, a, false, bb, (short)0, acc, false, false);
    }
    float sg = 0.f;
#pragma unroll
    for (int r = 0; r < 8; ++r) sg += acc[r];
    sg += 8.0f * b2f[ncol];
    const int srow = tm * 2 + ((lane >= 16) ? 1 : 0);
    sbuf[(size_t)srow * COUT + ncol] = sg;
}

// ---------------------------------------------------------------------------
// BN statistics: deterministic two-stage reduction (partial sums, no atomics)
// ---------------------------------------------------------------------------
__global__ void stats_f16_kernel(const _Float16* __restrict__ src,
                                 float* __restrict__ ps, float* __restrict__ ps2,
                                 int cout, int rows) {
    const int c = threadIdx.x, g = blockIdx.x, G = gridDim.x;
    float s = 0.f, s2 = 0.f;
    for (int r = g; r < rows; r += G) {
        float v = (float)src[(size_t)r * cout + c];
        s += v; s2 += v * v;
    }
    ps [(size_t)g * cout + c] = s;
    ps2[(size_t)g * cout + c] = s2;
}

__global__ void stats_f32relu_kernel(const float* __restrict__ src,
                                     float* __restrict__ ps, float* __restrict__ ps2,
                                     int cout, int rows) {
    const int c = threadIdx.x, g = blockIdx.x, G = gridDim.x;
    float s = 0.f, s2 = 0.f;
    for (int r = g; r < rows; r += G) {
        float v = src[(size_t)r * cout + c];
        v = v > 0.f ? v : 0.f;
        s += v; s2 += v * v;
    }
    ps [(size_t)g * cout + c] = s;
    ps2[(size_t)g * cout + c] = s2;
}

__global__ void finalize_bn_kernel(const float* __restrict__ ps,
                                   const float* __restrict__ ps2, int G,
                                   const float* __restrict__ gam,
                                   const float* __restrict__ bet,
                                   float* __restrict__ scale,
                                   float* __restrict__ shift,
                                   float invM, int cout) {
    const int c = threadIdx.x;
    if (c >= cout) return;
    float s = 0.f, s2 = 0.f;
    for (int i = 0; i < G; ++i) {
        s  += ps [(size_t)i * cout + c];
        s2 += ps2[(size_t)i * cout + c];
    }
    float mu  = s * invM;
    float var = s2 * invM - mu * mu;
    float inv = rsqrtf(var + BN_EPS);
    float sc  = gam[c] * inv;
    scale[c] = sc;
    shift[c] = bet[c] - mu * sc;
}

// BN2 apply on relu(s): writes next-layer features (f32 for kNN, f16 for WMMA)
__global__ void apply_bn2_kernel(const float* __restrict__ sbuf,
                                 const float* __restrict__ sc,
                                 const float* __restrict__ sh,
                                 float* __restrict__ xn,
                                 _Float16* __restrict__ xhn,
                                 int cout, int n) {
    int idx = blockIdx.x * blockDim.x + threadIdx.x;
    if (idx >= n) return;
    int c = idx % cout;
    float v = sbuf[idx];
    v = v > 0.f ? v : 0.f;
    v = sc[c] * v + sh[c];
    xn[idx]  = v;
    xhn[idx] = (_Float16)v;
}

// ---------------------------------------------------------------------------
// mean/max pool over P -> pooled [16, 512]
// ---------------------------------------------------------------------------
__global__ void pool_kernel(const float* __restrict__ xc,
                            float* __restrict__ pooled) {
    const int b = blockIdx.x;
    const int c = threadIdx.x;    // 0..255
    const float* base = xc + (size_t)b * PB * 256 + c;
    float s = 0.f, m = -3.4e38f;
    for (int p = 0; p < PB; ++p) {
        float v = base[(size_t)p * 256];
        s += v;
        if (v > m) m = v;
    }
    pooled[b * 512 + c]       = s * (1.f / PB);
    pooled[b * 512 + 256 + c] = m;
}

// head: Linear -> ReLU -> BN over the 16 rows (each thread owns a channel)
__global__ void head_fc_bn_kernel(const float* __restrict__ in,
                                  const float* __restrict__ w,
                                  const float* __restrict__ bias,
                                  const float* __restrict__ gam,
                                  const float* __restrict__ bet,
                                  float* __restrict__ out,
                                  int Cin, int Cout) {
    const int n = threadIdx.x;
    if (n >= Cout) return;
    float vals[BG];
    const float* wr = w + (size_t)n * Cin;
    for (int b = 0; b < BG; ++b) {
        float s = bias[n];
        const float* ir = in + (size_t)b * Cin;
        for (int k = 0; k < Cin; ++k) s += ir[k] * wr[k];
        vals[b] = s > 0.f ? s : 0.f;
    }
    float mu = 0.f;
    for (int b = 0; b < BG; ++b) mu += vals[b];
    mu *= (1.f / BG);
    float var = 0.f;
    for (int b = 0; b < BG; ++b) { float d = vals[b] - mu; var += d * d; }
    var *= (1.f / BG);
    float inv = rsqrtf(var + BN_EPS);
    for (int b = 0; b < BG; ++b)
        out[(size_t)b * Cout + n] = (vals[b] - mu) * inv * gam[n] + bet[n];
}

__global__ void head_last_kernel(const float* __restrict__ in,
                                 const float* __restrict__ w,
                                 const float* __restrict__ bias,
                                 float* __restrict__ out) {
    const int b = threadIdx.x;
    if (b >= BG) return;
    float lg[5];
    for (int j = 0; j < 5; ++j) {
        float s = bias[j];
        const float* wr = w + j * 128;
        const float* ir = in + b * 128;
        for (int k = 0; k < 128; ++k) s += ir[k] * wr[k];
        lg[j] = s;
    }
    float m = lg[0];
    for (int j = 1; j < 5; ++j) if (lg[j] > m) m = lg[j];
    float se = 0.f;
    for (int j = 0; j < 5; ++j) se += expf(lg[j] - m);
    float ls = m + logf(se);
    for (int j = 0; j < 5; ++j) out[b * 5 + j] = lg[j] - ls;
}

// ---------------------------------------------------------------------------
// host orchestration
// ---------------------------------------------------------------------------
extern "C" void kernel_launch(void* const* d_in, const int* in_sizes, int n_in,
                              void* d_out, int out_size, void* d_ws, size_t ws_size,
                              hipStream_t stream) {
    (void)n_in; (void)out_size; (void)ws_size;
    static const int CIN[4]  = {5, 64, 64, 128};
    static const int COUT[4] = {64, 64, 128, 256};

    const float *x = nullptr, *lw = nullptr, *lb = nullptr;
    const float *cw1[4], *cb1[4], *cg1[4], *cbt1[4];
    const float *cw2[4], *cb2[4], *cbng[4], *cbnb[4];
    const float *hw[2], *hb[2], *hg[2], *hbt[2];

    if (in_sizes[0] == NPTS * 5) {
        // insertion order: x, batch, convs{w1,b1,g1,bt1,w2,b2,bn_g,bn_b}x4,
        // hidden{w,b,g,bt}x2, last{w,b}
        x = (const float*)d_in[0];
        int p = 2;
        for (int l = 0; l < 4; ++l) {
            cw1[l]  = (const float*)d_in[p++]; cb1[l]  = (const float*)d_in[p++];
            cg1[l]  = (const float*)d_in[p++]; cbt1[l] = (const float*)d_in[p++];
            cw2[l]  = (const float*)d_in[p++]; cb2[l]  = (const float*)d_in[p++];
            cbng[l] = (const float*)d_in[p++]; cbnb[l] = (const float*)d_in[p++];
        }
        for (int l = 0; l < 2; ++l) {
            hw[l] = (const float*)d_in[p++]; hb[l]  = (const float*)d_in[p++];
            hg[l] = (const float*)d_in[p++]; hbt[l] = (const float*)d_in[p++];
        }
        lw = (const float*)d_in[p++]; lb = (const float*)d_in[p++];
    } else {
        // jax tree_flatten sorted-key order: batch, convs{b1,b2,bn_b,bn_g,bt1,g1,w1,w2}x4,
        // hidden{b,bt,g,w}x2, last{b,w}, x
        int p = 1;
        for (int l = 0; l < 4; ++l) {
            cb1[l]  = (const float*)d_in[p++]; cb2[l]  = (const float*)d_in[p++];
            cbnb[l] = (const float*)d_in[p++]; cbng[l] = (const float*)d_in[p++];
            cbt1[l] = (const float*)d_in[p++]; cg1[l]  = (const float*)d_in[p++];
            cw1[l]  = (const float*)d_in[p++]; cw2[l]  = (const float*)d_in[p++];
        }
        for (int l = 0; l < 2; ++l) {
            hb[l] = (const float*)d_in[p++]; hbt[l] = (const float*)d_in[p++];
            hg[l] = (const float*)d_in[p++]; hw[l]  = (const float*)d_in[p++];
        }
        lb = (const float*)d_in[p++]; lw = (const float*)d_in[p++];
        x  = (const float*)d_in[p++];
    }

    // workspace carving
    char* ws = (char*)d_ws;
    size_t off = 0;
    auto carve = [&](size_t bytes) -> void* {
        void* p = ws + off;
        off = (off + bytes + 255) & ~(size_t)255;
        return p;
    };
    float*    xc   = (float*)   carve((size_t)NPTS * 256 * 4);   // 16 MB
    _Float16* xh   = (_Float16*)carve((size_t)NPTS * 256 * 2);   //  8 MB
    int*      knn  = (int*)     carve((size_t)NPTS * K_NBR * 4);
    _Float16* h1h  = (_Float16*)carve((size_t)MEDGE * 256 * 2);  // 64 MB
    float*    sbuf = (float*)   carve((size_t)NPTS * 256 * 4);   // 16 MB
    float*    ps   = (float*)   carve((size_t)256 * 256 * 4);
    float*    ps2  = (float*)   carve((size_t)256 * 256 * 4);
    float*    sc1  = (float*)   carve(256 * 4);
    float*    sh1  = (float*)   carve(256 * 4);
    float*    sc2  = (float*)   carve(256 * 4);
    float*    sh2  = (float*)   carve(256 * 4);
    _Float16* w1h[4]; _Float16* w2h[4];
    for (int l = 0; l < 4; ++l) {
        w1h[l] = (_Float16*)carve((size_t)COUT[l] * 2 * CIN[l] * 2);
        w2h[l] = (_Float16*)carve((size_t)COUT[l] * COUT[l] * 2);
    }
    float* pooled = (float*)carve(BG * 512 * 4);
    float* hbuf1  = (float*)carve(BG * 256 * 4);
    float* hbuf2  = (float*)carve(BG * 128 * 4);

    // weight conversion to f16
    for (int l = 0; l < 4; ++l) {
        int n1 = COUT[l] * 2 * CIN[l];
        f32_to_f16_kernel<<<(n1 + 255) / 256, 256, 0, stream>>>(cw1[l], w1h[l], n1);
        int n2 = COUT[l] * COUT[l];
        f32_to_f16_kernel<<<(n2 + 255) / 256, 256, 0, stream>>>(cw2[l], w2h[l], n2);
    }

    // init features
    {
        int n = NPTS * 5;
        copy_x_kernel<<<(n + 255) / 256, 256, 0, stream>>>(x, xc, xh, n);
    }

    // EdgeConv layers
    for (int l = 0; l < 4; ++l) {
        const int cin = CIN[l], cout = COUT[l];
        knn_kernel<<<NPTS, 32, 0, stream>>>(xc, knn, cin);

        dim3 g(MEDGE / 16 / 4, cout / 16);   // 4 waves per block
        switch (l) {
        case 0: gemm1_edge_kernel<5, 64 ><<<g, 128, 0, stream>>>(xh, knn, w1h[l], cb1[l], h1h); break;
        case 1: gemm1_edge_kernel<64, 64 ><<<g, 128, 0, stream>>>(xh, knn, w1h[l], cb1[l], h1h); break;
        case 2: gemm1_edge_kernel<64, 128><<<g, 128, 0, stream>>>(xh, knn, w1h[l], cb1[l], h1h); break;
        case 3: gemm1_edge_kernel<128, 256><<<g, 128, 0, stream>>>(xh, knn, w1h[l], cb1[l], h1h); break;
        }

        stats_f16_kernel<<<256, cout, 0, stream>>>(h1h, ps, ps2, cout, MEDGE);
        finalize_bn_kernel<<<1, cout, 0, stream>>>(ps, ps2, 256, cg1[l], cbt1[l],
                                                   sc1, sh1, 1.0f / MEDGE, cout);
        {
            int n = MEDGE * cout;
            bn1_apply_kernel<<<(n + 255) / 256, 256, 0, stream>>>(h1h, sc1, sh1, cout, n);
        }

        switch (l) {
        case 0: gemm2_fused_kernel<64 ><<<g, 128, 0, stream>>>(h1h, w2h[l], cb2[l], sbuf); break;
        case 1: gemm2_fused_kernel<64 ><<<g, 128, 0, stream>>>(h1h, w2h[l], cb2[l], sbuf); break;
        case 2: gemm2_fused_kernel<128><<<g, 128, 0, stream>>>(h1h, w2h[l], cb2[l], sbuf); break;
        case 3: gemm2_fused_kernel<256><<<g, 128, 0, stream>>>(h1h, w2h[l], cb2[l], sbuf); break;
        }

        stats_f32relu_kernel<<<256, cout, 0, stream>>>(sbuf, ps, ps2, cout, NPTS);
        finalize_bn_kernel<<<1, cout, 0, stream>>>(ps, ps2, 256, cbng[l], cbnb[l],
                                                   sc2, sh2, 1.0f / NPTS, cout);

        int n = NPTS * cout;
        apply_bn2_kernel<<<(n + 255) / 256, 256, 0, stream>>>(sbuf, sc2, sh2, xc, xh, cout, n);
    }

    // pooling + head
    pool_kernel<<<BG, 256, 0, stream>>>(xc, pooled);
    head_fc_bn_kernel<<<1, 256, 0, stream>>>(pooled, hw[0], hb[0], hg[0], hbt[0], hbuf1, 512, 256);
    head_fc_bn_kernel<<<1, 128, 0, stream>>>(hbuf1,  hw[1], hb[1], hg[1], hbt[1], hbuf2, 256, 128);
    head_last_kernel<<<1, 32, 0, stream>>>(hbuf2, lw, lb, (float*)d_out);
}
